// MultiHeadHomogeneousAttention_45165876084976
// MI455X (gfx1250) — matrix operand order repaired and checked by reading, not compiled, not run on hardware
//
#include <hip/hip_runtime.h>
#include <hip/hip_bf16.h>

// ---------------------------------------------------------------------------
// MultiHeadHomogeneousAttention for MI455X (gfx1250), bf16 WMMA pipeline.
//   B=4, S=2048, D=1024, H=8, P=128, KTAPS=3 (causal conv taps on K/V).
// GEMMs on v_wmma_f32_16x16x32_bf16, software-pipelined; B-tiles of the two
// dense GEMMs staged through LDS by the Tensor Data Mover (TENSORcnt path).
// ---------------------------------------------------------------------------

typedef __bf16 bf16;
typedef __attribute__((ext_vector_type(16))) __bf16 bf16x16;
typedef __attribute__((ext_vector_type(8)))  __bf16 bf16x8;
typedef __attribute__((ext_vector_type(8)))  float  f32x8;
typedef __attribute__((ext_vector_type(4)))  unsigned int u32x4;
typedef __attribute__((ext_vector_type(4)))  int    i32x4;
typedef __attribute__((ext_vector_type(8)))  int    i32x8;

#define DIM_M    1024
#define DIM_PROJ 128
#define NHEAD    8
#define SEQ      2048
#define BATCH    4
#define KTAPS    3

#if defined(__has_builtin)
#if __has_builtin(__builtin_amdgcn_tensor_load_to_lds)
#define HAS_TDM 1
#endif
#if __has_builtin(__builtin_amdgcn_s_wait_tensorcnt)
#define HAS_TWAIT 1
#endif
#if __has_builtin(__builtin_amdgcn_sched_barrier)
#define HAS_SCHEDB 1
#endif
#endif
#ifndef HAS_TDM
#define HAS_TDM 0
#endif
#ifndef HAS_TWAIT
#define HAS_TWAIT 0
#endif
#ifndef HAS_SCHEDB
#define HAS_SCHEDB 0
#endif

#if __has_include(<hip/amd_detail/amd_gfx1250_TDM.h>)
#define TDM_SIX 1   // therock headers -> 6-arg builtin
#else
#define TDM_SIX 0   // ROCm 7.2 -> 5-arg builtin
#endif

// Scheduler fence: keep bulk loads above, WMMA blocks below.
static __device__ __forceinline__ void sched_fence() {
#if HAS_SCHEDB
  __builtin_amdgcn_sched_barrier(0);
#endif
}

// ----------------------------- fragment helpers ----------------------------
// A-fragment (16x32 bf16, MxK). Lane l: row = l&15, half = l>>4.
static __device__ __forceinline__ bf16x16
frag_a_load(const bf16* base, int lda, int lane) {
  const int row = lane & 15, half = lane >> 4;
  const bf16* p = base + (size_t)row * lda + 8 * half;
  bf16x8 lo = *(const bf16x8*)(p);
  bf16x8 hi = *(const bf16x8*)(p + 16);
  bf16x16 r;
#pragma unroll
  for (int i = 0; i < 8; ++i) { r[i] = lo[i]; r[i + 8] = hi[i]; }
  return r;
}

// B-fragment (32x16 bf16, KxN) from an N-major source: element (k,n) at
// base[n*ldb + k]. Lane l: col = l&15, K runs at 8*half and 16+8*half.
static __device__ __forceinline__ bf16x16
frag_b_load(const bf16* base, int ldb, int lane) {
  const int col = lane & 15, half = lane >> 4;
  const bf16* p = base + (size_t)col * ldb + 8 * half;
  bf16x8 lo = *(const bf16x8*)(p);
  bf16x8 hi = *(const bf16x8*)(p + 16);
  bf16x16 r;
#pragma unroll
  for (int i = 0; i < 8; ++i) { r[i] = lo[i]; r[i + 8] = hi[i]; }
  return r;
}

// A-fragment with per-lane causal sequence shift (zeros before seq start).
static __device__ __forceinline__ bf16x16
frag_a_load_shift(const bf16* X, int m, int shift, int k, int lane) {
  const int half = lane >> 4;
  const int s = m & (SEQ - 1), b = m >> 11;
  const int ss = s + shift;
  bf16x16 r;
  if (ss >= 0) {
    const bf16* p = X + ((size_t)(b * SEQ + ss)) * DIM_M + k + 8 * half;
    bf16x8 lo = *(const bf16x8*)(p);
    bf16x8 hi = *(const bf16x8*)(p + 16);
#pragma unroll
    for (int i = 0; i < 8; ++i) { r[i] = lo[i]; r[i + 8] = hi[i]; }
  } else {
#pragma unroll
    for (int i = 0; i < 16; ++i) r[i] = (bf16)0.0f;
  }
  return r;
}

static __device__ __forceinline__ f32x8 zero8() {
  f32x8 z;
#pragma unroll
  for (int i = 0; i < 8; ++i) z[i] = 0.0f;
  return z;
}

static __device__ __forceinline__ f32x8
wmma_bf16(bf16x16 a, bf16x16 b, f32x8 c) {
  return __builtin_amdgcn_wmma_f32_16x16x32_bf16(
      false, a, false, b, (short)0, c, false, false);
}

// 8 B-fragments covering a 128-wide N strip at one K block.
struct BTile { bf16x16 v[8]; };
static __device__ __forceinline__ BTile
load_btile(const bf16* base, int ldm, int lane) {
  BTile t;
#pragma unroll
  for (int j = 0; j < 8; ++j)
    t.v[j] = frag_b_load(base + (size_t)(j * 16) * ldm, ldm, lane);
  return t;
}

// ------------------------ TDM staging of B tiles ---------------------------
// Stage a 128-row x 32-element bf16 tile (row stride DIM_M elements) from
// global into LDS (packed n-major, ldb=32). Issued by wave 0 only.
static __device__ __forceinline__ void
stage_b_tile(bf16* ldst, const bf16* gsrc, int tid) {
#if HAS_TDM
  if ((tid >> 5) == 0) {
    const unsigned long long ga = (unsigned long long)(size_t)gsrc;
    const unsigned lds_off = (unsigned)(size_t)(void*)ldst;
    u32x4 g0;
    g0[0] = 1u;                                          // count=1 (valid)
    g0[1] = lds_off;                                     // lds_addr
    g0[2] = (unsigned)ga;                                // global_addr lo
    g0[3] = (unsigned)((ga >> 32) & 0x01FFFFFFu) | (2u << 30);  // hi | type=2
    i32x8 g1;
    g1[0] = (int)(1u << 16);                             // data_size = 2B
    g1[1] = (int)(((unsigned)DIM_M & 0xFFFFu) << 16);    // tensor_dim0 lo16
    g1[2] = (int)(((unsigned)DIM_M >> 16) | ((4096u & 0xFFFFu) << 16));
    g1[3] = (int)((4096u >> 16) | (32u << 16));          // dim1 hi | tile_dim0
    g1[4] = (int)(128u);                                 // tile_dim1=128
    g1[5] = (int)((unsigned)DIM_M);                      // dim0_stride lo32
    g1[6] = 0;
    g1[7] = 0;
    i32x4 z4; z4[0] = 0; z4[1] = 0; z4[2] = 0; z4[3] = 0;
#if TDM_SIX
    i32x8 z8;
#pragma unroll
    for (int i = 0; i < 8; ++i) z8[i] = 0;
    __builtin_amdgcn_tensor_load_to_lds(g0, g1, z4, z4, z8, 0);
#else
    __builtin_amdgcn_tensor_load_to_lds(g0, g1, z4, z4, 0);
#endif
  }
#else
  // Fallback: cooperative copy, 256 threads x 16 bf16.
  const int r = tid >> 1, c = (tid & 1) * 16;
  *(bf16x16*)(ldst + r * 32 + c) = *(const bf16x16*)(gsrc + (size_t)r * DIM_M + c);
#endif
}

static __device__ __forceinline__ void publish_tile() {
#if HAS_TDM && HAS_TWAIT
  if (threadIdx.x < 32) __builtin_amdgcn_s_wait_tensorcnt(0);
#endif
  __syncthreads();
}

// ------------------------------- prep kernels ------------------------------
__global__ void __launch_bounds__(256)
cvt_kernel(const float* __restrict__ in, bf16* __restrict__ out, int n) {
  for (int i = blockIdx.x * blockDim.x + threadIdx.x; i < n;
       i += gridDim.x * blockDim.x)
    out[i] = (bf16)in[i];
}

__global__ void __launch_bounds__(256)
pack_tap_kernel(const float* __restrict__ in, bf16* __restrict__ out, int nd) {
  const int total = nd * KTAPS;
  for (int i = blockIdx.x * blockDim.x + threadIdx.x; i < total;
       i += gridDim.x * blockDim.x) {
    const int t = i / nd;
    const int e = i - t * nd;
    out[i] = (bf16)in[(size_t)e * KTAPS + t];
  }
}

// ----------------- TDM-staged GEMM body (shared by Q / out) ----------------
// Computes acc[0..7] (16 rows x 128 cols) for one wave. B tiles staged by
// TDM into double-buffered LDS; A fragments double-buffered in registers.
// Manual 2x unroll keeps all buffer indices compile-time constant.
static __device__ __forceinline__ void
gemm_tdm_body(f32x8* acc, const bf16* Xrow, const bf16* Wc,
              bf16 (*bstage)[128 * 32], int lane, int tid) {
  stage_b_tile(&bstage[0][0], Wc, tid);
  bf16x16 a0 = frag_a_load(Xrow, DIM_M, lane), a1;
  publish_tile();
  for (int t = 0; t < DIM_M / 32; t += 2) {
    // stage tile t+1 into buf1, prefetch A for t+1
    stage_b_tile(&bstage[1][0], Wc + (t + 1) * 32, tid);
    a1 = frag_a_load(Xrow + (t + 1) * 32, DIM_M, lane);
    {
      BTile bt = load_btile(&bstage[0][0], 32, lane);   // 16 ds_loads
      sched_fence();                                    // keep loads above
#pragma unroll
      for (int j = 0; j < 8; ++j) acc[j] = wmma_bf16(a0, bt.v[j], acc[j]);
    }
    publish_tile();   // buf1 ready; all waves done reading buf0
    if (t + 2 < DIM_M / 32) {
      stage_b_tile(&bstage[0][0], Wc + (t + 2) * 32, tid);
      a0 = frag_a_load(Xrow + (t + 2) * 32, DIM_M, lane);
    }
    {
      BTile bt = load_btile(&bstage[1][0], 32, lane);
      sched_fence();
#pragma unroll
      for (int j = 0; j < 8; ++j) acc[j] = wmma_bf16(a1, bt.v[j], acc[j]);
    }
    publish_tile();   // buf0 ready; all waves done reading buf1
  }
}

// ------------------ Q projection GEMM (TDM-staged B tiles) -----------------
__global__ void __launch_bounds__(256)
proj_q_kernel(const bf16* __restrict__ X, const bf16* __restrict__ W,
              const float* __restrict__ bias, bf16* __restrict__ out) {
  __shared__ bf16 bstage[2][128 * 32];
  const int lane = threadIdx.x & 31, wave = threadIdx.x >> 5;
  const int row0 = blockIdx.x * 128 + wave * 16;
  const int col0 = blockIdx.y * 128;
  f32x8 acc[8];
#pragma unroll
  for (int j = 0; j < 8; ++j) acc[j] = zero8();

  gemm_tdm_body(acc, X + (size_t)row0 * DIM_M, W + (size_t)col0 * DIM_M,
                bstage, lane, threadIdx.x);

  const int half = lane >> 4, cl = lane & 15;
#pragma unroll
  for (int j = 0; j < 8; ++j) {
    const int n = col0 + j * 16 + cl;
    const float bi = bias[n];
    const int h = n >> 7, p = n & (DIM_PROJ - 1);
#pragma unroll
    for (int r = 0; r < 8; ++r) {
      const int m = row0 + r + 8 * half;
      const int b = m >> 11, s = m & (SEQ - 1);
      out[(((size_t)b * NHEAD + h) * SEQ + s) * DIM_PROJ + p] =
          (bf16)(acc[j][r] + bi);
    }
  }
}

// ---------- K/V projection (3-tap causal conv, register-pipelined) ---------
__global__ void __launch_bounds__(256)
proj_conv_kernel(const bf16* __restrict__ X, const bf16* __restrict__ Wt,
                 const float* __restrict__ bias, bf16* __restrict__ out,
                 int transposed) {
  const int lane = threadIdx.x & 31, wave = threadIdx.x >> 5;
  const int row0 = blockIdx.x * 128 + wave * 16;
  const int col0 = blockIdx.y * 128;
  const int mlane = row0 + (lane & 15);
  f32x8 acc[8];
#pragma unroll
  for (int j = 0; j < 8; ++j) acc[j] = zero8();

  const int NSTEP = KTAPS * (DIM_M / 32);   // 96
  auto stepA = [&](int st) {
    const int t = st >> 5, k = (st & 31) << 5;
    return frag_a_load_shift(X, mlane, t - (KTAPS - 1), k, lane);
  };
  auto stepB = [&](int st) {
    const int t = st >> 5, k = (st & 31) << 5;
    return load_btile(Wt + (size_t)t * DIM_M * DIM_M + (size_t)col0 * DIM_M + k,
                      DIM_M, lane);
  };
  bf16x16 a0 = stepA(0), a1;
  BTile b0 = stepB(0), b1;
  for (int st = 0; st < NSTEP; st += 2) {
    a1 = stepA(st + 1);
    b1 = stepB(st + 1);
    sched_fence();    // next tile's loads stay above this WMMA block
#pragma unroll
    for (int j = 0; j < 8; ++j) acc[j] = wmma_bf16(a0, b0.v[j], acc[j]);
    if (st + 2 < NSTEP) { a0 = stepA(st + 2); b0 = stepB(st + 2); }
    sched_fence();
#pragma unroll
    for (int j = 0; j < 8; ++j) acc[j] = wmma_bf16(a1, b1.v[j], acc[j]);
  }

  const int half = lane >> 4, cl = lane & 15;
#pragma unroll
  for (int j = 0; j < 8; ++j) {
    const int n = col0 + j * 16 + cl;
    const float bi = bias[n];
    const int h = n >> 7, p = n & (DIM_PROJ - 1);
#pragma unroll
    for (int r = 0; r < 8; ++r) {
      const int m = row0 + r + 8 * half;
      const int bb = m >> 11, s = m & (SEQ - 1);
      const float v = acc[j][r] + bi;
      if (transposed)
        out[(((size_t)bb * NHEAD + h) * DIM_PROJ + p) * SEQ + s] = (bf16)v;
      else
        out[(((size_t)bb * NHEAD + h) * SEQ + s) * DIM_PROJ + p] = (bf16)v;
    }
  }
}

// ----------------------- flash attention (per b,h) -------------------------
__global__ void __launch_bounds__(256)
attn_kernel(const bf16* __restrict__ qb, const bf16* __restrict__ kb,
            const bf16* __restrict__ vT, bf16* __restrict__ ctx) {
  __shared__ bf16 smem[8][16 * 32];   // per-wave probability tile
  const int lane = threadIdx.x & 31, wave = threadIdx.x >> 5;
  const int half = lane >> 4, cl = lane & 15;
  const int bh = blockIdx.y;
  const int q0 = blockIdx.x * 128 + wave * 16;
  const bf16* qh = qb + (size_t)bh * SEQ * DIM_PROJ;
  const bf16* kh = kb + (size_t)bh * SEQ * DIM_PROJ;
  const bf16* vh = vT + (size_t)bh * DIM_PROJ * SEQ;

  bf16x16 qa[4];
#pragma unroll
  for (int ks = 0; ks < 4; ++ks)
    qa[ks] = frag_a_load(qh + (size_t)q0 * DIM_PROJ + ks * 32, DIM_PROJ, lane);

  f32x8 acc[8];
#pragma unroll
  for (int j = 0; j < 8; ++j) acc[j] = zero8();
  float mrow[8], lrow[8];
#pragma unroll
  for (int r = 0; r < 8; ++r) { mrow[r] = -3.0e38f; lrow[r] = 0.0f; }

  const float scale = 0.08838834764831845f;  // P^{-1/2}
  bf16* my = &smem[wave][0];

  for (int k0 = 0; k0 < SEQ; k0 += 32) {
    // all 8 key fragments first, then the score WMMAs back-to-back
    bf16x16 kf[8];
#pragma unroll
    for (int ks = 0; ks < 4; ++ks) {
      kf[ks]     = frag_b_load(kh + (size_t)k0 * DIM_PROJ + ks * 32,
                               DIM_PROJ, lane);
      kf[ks + 4] = frag_b_load(kh + (size_t)(k0 + 16) * DIM_PROJ + ks * 32,
                               DIM_PROJ, lane);
    }
    sched_fence();
    f32x8 s0 = zero8(), s1 = zero8();
#pragma unroll
    for (int ks = 0; ks < 4; ++ks) s0 = wmma_bf16(qa[ks], kf[ks], s0);
#pragma unroll
    for (int ks = 0; ks < 4; ++ks) s1 = wmma_bf16(qa[ks], kf[ks + 4], s1);

    // V fragments in flight behind the softmax VALU work
    BTile vt = load_btile(vh + k0, SEQ, lane);
    sched_fence();

#pragma unroll
    for (int r = 0; r < 8; ++r) {
      float v0 = s0[r] * scale, v1 = s1[r] * scale;
      float mx = fmaxf(v0, v1);
#pragma unroll
      for (int off = 1; off < 16; off <<= 1)
        mx = fmaxf(mx, __shfl_xor(mx, off, 32));
      const float mnew = fmaxf(mrow[r], mx);
      const float alpha = __expf(mrow[r] - mnew);
      const float p0 = __expf(v0 - mnew), p1 = __expf(v1 - mnew);
      float ps = p0 + p1;
#pragma unroll
      for (int off = 1; off < 16; off <<= 1) ps += __shfl_xor(ps, off, 32);
      lrow[r] = lrow[r] * alpha + ps;
      mrow[r] = mnew;
#pragma unroll
      for (int j = 0; j < 8; ++j) acc[j][r] *= alpha;
      my[(r + 8 * half) * 32 + cl] = (bf16)p0;
      my[(r + 8 * half) * 32 + 16 + cl] = (bf16)p1;
    }
    // wave-private LDS: stores -> load ordered by in-order DScnt, no barrier
    asm volatile("s_wait_dscnt 0x0" ::: "memory");
    bf16x16 pa = frag_a_load(my, 32, lane);
#pragma unroll
    for (int j = 0; j < 8; ++j) acc[j] = wmma_bf16(pa, vt.v[j], acc[j]);
  }

  const int b = bh >> 3, h = bh & (NHEAD - 1);
#pragma unroll
  for (int j = 0; j < 8; ++j) {
#pragma unroll
    for (int r = 0; r < 8; ++r) {
      const float val = acc[j][r] / lrow[r];
      const int srow = q0 + r + 8 * half;
      ctx[((size_t)b * SEQ + srow) * DIM_M + h * DIM_PROJ + j * 16 + cl] =
          (bf16)val;
    }
  }
}

// --------- output projection + bias + residual (TDM-staged B tiles) --------
__global__ void __launch_bounds__(256)
out_gemm_kernel(const bf16* __restrict__ A, const bf16* __restrict__ W,
                const float* __restrict__ bo, const float* __restrict__ resid,
                float* __restrict__ Hout) {
  __shared__ bf16 bstage[2][128 * 32];
  const int lane = threadIdx.x & 31, wave = threadIdx.x >> 5;
  const int row0 = blockIdx.x * 128 + wave * 16;
  const int col0 = blockIdx.y * 128;
  f32x8 acc[8];
#pragma unroll
  for (int j = 0; j < 8; ++j) acc[j] = zero8();

  gemm_tdm_body(acc, A + (size_t)row0 * DIM_M, W + (size_t)col0 * DIM_M,
                bstage, lane, threadIdx.x);

  const int half = lane >> 4, cl = lane & 15;
#pragma unroll
  for (int j = 0; j < 8; ++j) {
    const int n = col0 + j * 16 + cl;
    const float bi = bo[n];
#pragma unroll
    for (int r = 0; r < 8; ++r) {
      const int m = row0 + r + 8 * half;
      Hout[(size_t)m * DIM_M + n] =
          acc[j][r] + bi + resid[(size_t)m * DIM_M + n];
    }
  }
}

// ------------------------------ LayerNorm ----------------------------------
__global__ void __launch_bounds__(256)
ln_kernel(const float* __restrict__ H, const float* __restrict__ gamma,
          const float* __restrict__ beta, float* __restrict__ out) {
  const int row = blockIdx.x;
  const float* hr = H + (size_t)row * DIM_M;
  float s = 0.0f, s2 = 0.0f;
  for (int i = threadIdx.x; i < DIM_M; i += 256) {
    const float v = hr[i];
    s += v; s2 += v * v;
  }
#pragma unroll
  for (int off = 16; off >= 1; off >>= 1) {
    s += __shfl_xor(s, off, 32);
    s2 += __shfl_xor(s2, off, 32);
  }
  __shared__ float ss[8], ss2[8];
  const int lane = threadIdx.x & 31, wave = threadIdx.x >> 5;
  if (lane == 0) { ss[wave] = s; ss2[wave] = s2; }
  __syncthreads();
  s = 0.0f; s2 = 0.0f;
#pragma unroll
  for (int w = 0; w < 8; ++w) { s += ss[w]; s2 += ss2[w]; }
  const float mu = s * (1.0f / DIM_M);
  const float var = s2 * (1.0f / DIM_M) - mu * mu;
  const float inv = rsqrtf(var + 1e-12f);
  for (int i = threadIdx.x; i < DIM_M; i += 256)
    out[(size_t)row * DIM_M + i] = (hr[i] - mu) * inv * gamma[i] + beta[i];
}

// ------------------------------- launcher ----------------------------------
extern "C" void kernel_launch(void* const* d_in, const int* in_sizes, int n_in,
                              void* d_out, int out_size, void* d_ws,
                              size_t ws_size, hipStream_t stream) {
  const float* value = (const float*)d_in[0];
  const float* keyx  = (const float*)d_in[1];
  const float* query = (const float*)d_in[2];
  const float* Wq    = (const float*)d_in[3];
  const float* bq    = (const float*)d_in[4];
  const float* Wk    = (const float*)d_in[5];
  const float* bk    = (const float*)d_in[6];
  const float* Wv    = (const float*)d_in[7];
  const float* bv    = (const float*)d_in[8];
  const float* Wo    = (const float*)d_in[9];
  const float* bo    = (const float*)d_in[10];
  const float* gamma = (const float*)d_in[11];
  const float* beta  = (const float*)d_in[12];
  float* out = (float*)d_out;

  const size_t NE_X = (size_t)BATCH * SEQ * DIM_M;        // 8388608
  const size_t NE_W = (size_t)NHEAD * DIM_PROJ * DIM_M;   // 1048576

  char* ws = (char*)d_ws;
  bf16* xq  = (bf16*)ws;  ws += NE_X * 2;
  bf16* xk  = (bf16*)ws;  ws += NE_X * 2;
  bf16* xv  = (bf16*)ws;  ws += NE_X * 2;
  bf16* wqb = (bf16*)ws;  ws += NE_W * 2;
  bf16* wkb = (bf16*)ws;  ws += NE_W * KTAPS * 2;
  bf16* wvb = (bf16*)ws;  ws += NE_W * KTAPS * 2;
  bf16* wob = (bf16*)ws;  ws += NE_W * 2;
  bf16* qbf = (bf16*)ws;  ws += NE_X * 2;   // (B,H,S,P)
  bf16* kbf = (bf16*)ws;  ws += NE_X * 2;   // (B,H,S,P)
  bf16* vTt = (bf16*)ws;  ws += NE_X * 2;   // (B,H,P,S)
  bf16* ctx = (bf16*)ws;  ws += NE_X * 2;   // (B,S,H*P)
  float* hb = (float*)ws; ws += NE_X * 4;   // residual+bias, pre-LN

  dim3 blk(256);
  cvt_kernel<<<2048, blk, 0, stream>>>(query, xq, (int)NE_X);
  cvt_kernel<<<2048, blk, 0, stream>>>(keyx,  xk, (int)NE_X);
  cvt_kernel<<<2048, blk, 0, stream>>>(value, xv, (int)NE_X);
  cvt_kernel<<<512, blk, 0, stream>>>(Wq, wqb, (int)NE_W);
  cvt_kernel<<<512, blk, 0, stream>>>(Wo, wob, (int)NE_W);
  pack_tap_kernel<<<1024, blk, 0, stream>>>(Wk, wkb, (int)NE_W);
  pack_tap_kernel<<<1024, blk, 0, stream>>>(Wv, wvb, (int)NE_W);

  dim3 gproj(BATCH * SEQ / 128, DIM_M / 128);   // (64, 8)
  proj_q_kernel<<<gproj, blk, 0, stream>>>(xq, wqb, bq, qbf);
  proj_conv_kernel<<<gproj, blk, 0, stream>>>(xk, wkb, bk, kbf, 0);
  proj_conv_kernel<<<gproj, blk, 0, stream>>>(xv, wvb, bv, vTt, 1);

  dim3 gattn(SEQ / 128, BATCH * NHEAD);         // (16, 32)
  attn_kernel<<<gattn, blk, 0, stream>>>(qbf, kbf, vTt, ctx);

  out_gemm_kernel<<<gproj, blk, 0, stream>>>(ctx, wob, bo, query, hb);
  ln_kernel<<<BATCH * SEQ, blk, 0, stream>>>(hb, gamma, beta, out);
}
